// GCN_large_20564303414147
// MI455X (gfx1250) — matrix operand order repaired and checked by reading, not compiled, not run on hardware
//
#include <hip/hip_runtime.h>
#include <hip/hip_bf16.h>
#include <math.h>

typedef __attribute__((ext_vector_type(16))) _Float16 v16h;
typedef __attribute__((ext_vector_type(8)))  _Float16 v8h;
typedef __attribute__((ext_vector_type(4)))  _Float16 v4h;
typedef __attribute__((ext_vector_type(8)))  float    v8f;
typedef __attribute__((ext_vector_type(4)))  float    v4f;

#define GNODE 112
#define MT    7        // 112/16 output-row tiles
#define XP    136      // xs pitch in halves (128 used + 8 pad -> bank spread)
#define HP    136      // hT pitch in halves
#define APCH  264      // ag pitch in halves (256 used + 8 pad)
#define NWAVES 8

__device__ __forceinline__ float relu1(float v) {
  // single v_med3_f32: median(x, 0, +inf) == max(x, 0)
  return __builtin_amdgcn_fmed3f(v, 0.0f, __builtin_inff());
}

__device__ __forceinline__ v16h join16(v8h lo, v8h hi) {
  return __builtin_shufflevector(lo, hi, 0,1,2,3,4,5,6,7,8,9,10,11,12,13,14,15);
}

// A fragment (16x32 f16) from row-major [M][pitch]:
// lanes 0-15: M=mt*16+lane, K = kb*32 + 0..15
// lanes16-31: M=mt*16+lane-16, K = kb*32 + 16..31
__device__ __forceinline__ v16h loadA(const _Float16* base, int pitch, int mt, int kb, int lane) {
  const _Float16* p = base + (mt*16 + (lane & 15)) * pitch + kb*32 + (lane & 16);
  v8h lo = *(const v8h*)p;        // 16B aligned: pitch*2 and kb*64 are multiples of 16
  v8h hi = *(const v8h*)(p + 8);
  return join16(lo, hi);
}

// B fragment (32x16 f16) from B^T stored row-major [N][pitch]:
// lanes 0-15: N=nt*16+lane, K=kb*32+0..15 ; lanes 16-31: same N, K=+16..31
__device__ __forceinline__ v16h loadB_lds(const _Float16* base, int pitch, int nt, int kb, int lane) {
  const _Float16* p = base + (nt*16 + (lane & 15)) * pitch + kb*32 + (lane & 16);
  v8h lo = *(const v8h*)p;
  v8h hi = *(const v8h*)(p + 8);
  return join16(lo, hi);
}
__device__ __forceinline__ v16h loadB_glb(const _Float16* base, int pitch, int nt, int kb, int lane) {
  // global weight rows are 128/256 halves -> 32B-aligned v16h load
  return *(const v16h*)(base + (nt*16 + (lane & 15)) * pitch + kb*32 + (lane & 16));
}

// ag[n][d] = sum_m xs[n][m] * h[m][d]; h given transposed as hT[d][m]; K = 128 (m padded).
// Each wave computes a 16x32 strip: one A fragment feeds two WMMAs.
__device__ void agg_stage(const _Float16* xs, const _Float16* hT, _Float16* ag,
                          int D, int wave, int lane) {
  const int nP = D >> 5;               // pairs of N tiles
  const int nwork = MT * nP;
  for (int t = wave; t < nwork; t += NWAVES) {
    int mt = t / nP, np = t - mt * nP;
    int nt0 = np * 2, nt1 = nt0 + 1;
    v8f c0 = {}, c1 = {};
#pragma unroll
    for (int kb = 0; kb < 4; ++kb) {
      v16h a  = loadA(xs, XP, mt, kb, lane);
      v16h b0 = loadB_lds(hT, HP, nt0, kb, lane);
      v16h b1 = loadB_lds(hT, HP, nt1, kb, lane);
      c0 = __builtin_amdgcn_wmma_f32_16x16x32_f16(false, a, false, b0, (short)0, c0, false, false);
      c1 = __builtin_amdgcn_wmma_f32_16x16x32_f16(false, a, false, b1, (short)0, c1, false, false);
    }
    int dcol0 = nt0*16 + (lane & 15);
    int rbase = mt*16 + ((lane & 16) >> 1);
#pragma unroll
    for (int r = 0; r < 8; ++r) {
      ag[(rbase + r) * APCH + dcol0]      = (_Float16)c0[r];
      ag[(rbase + r) * APCH + dcol0 + 16] = (_Float16)c1[r];
    }
  }
}

// hT[d][n] = relu( sum_k A[n][k] * W[d][k] + bias[d] )  (W is N x K row-major == B^T).
// Bias pre-loaded into the accumulator; 16x32 strip per wave (A fragment reuse x2).
__device__ void proj_stage(const _Float16* A, int apitch, int K,
                           const _Float16* __restrict__ W, const float* __restrict__ bias,
                           int N, _Float16* hT, int wave, int lane) {
  const int nP = N >> 5;
  const int nwork = MT * nP;
  const int kblocks = K >> 5;
  for (int t = wave; t < nwork; t += NWAVES) {
    int mt = t / nP, np = t - mt * nP;
    int nt0 = np * 2, nt1 = nt0 + 1;
    int dcol0 = nt0*16 + (lane & 15);
    int dcol1 = dcol0 + 16;
    float bv0 = bias[dcol0], bv1 = bias[dcol1];
    v8f c0 = {bv0,bv0,bv0,bv0,bv0,bv0,bv0,bv0};
    v8f c1 = {bv1,bv1,bv1,bv1,bv1,bv1,bv1,bv1};
    for (int kb = 0; kb < kblocks; ++kb) {
      v16h a  = loadA(A, apitch, mt, kb, lane);
      v16h b0 = loadB_glb(W, K, nt0, kb, lane);
      v16h b1 = loadB_glb(W, K, nt1, kb, lane);
      c0 = __builtin_amdgcn_wmma_f32_16x16x32_f16(false, a, false, b0, (short)0, c0, false, false);
      c1 = __builtin_amdgcn_wmma_f32_16x16x32_f16(false, a, false, b1, (short)0, c1, false, false);
    }
    v8h o0, o1;
#pragma unroll
    for (int r = 0; r < 8; ++r) {
      o0[r] = (_Float16)relu1(c0[r]);
      o1[r] = (_Float16)relu1(c1[r]);
    }
    // transposed store: 8 consecutive node-rows for fixed d -> one 16B LDS store each
    int mbase = mt*16 + ((lane & 16) >> 1);
    *(v8h*)(hT + dcol0 * HP + mbase) = o0;
    *(v8h*)(hT + dcol1 * HP + mbase) = o1;
  }
}

__global__ void __launch_bounds__(256) cvt_pad(const float* __restrict__ src,
                                               _Float16* __restrict__ dst,
                                               int rows, int scols, int dcols) {
  int i = blockIdx.x * blockDim.x + threadIdx.x;
  if (i >= rows * dcols) return;
  int r = i / dcols, c = i - r * dcols;
  dst[i] = (c < scols) ? (_Float16)src[r * scols + c] : (_Float16)0.f;
}

__global__ void __launch_bounds__(256) gcn_main(
    const float* __restrict__ x,
    const _Float16* __restrict__ W1p, const float* __restrict__ b1,
    const _Float16* __restrict__ W2p, const float* __restrict__ b2,
    const _Float16* __restrict__ W3p, const float* __restrict__ b3,
    const _Float16* __restrict__ W4p, const float* __restrict__ b4,
    const _Float16* __restrict__ W5p, const float* __restrict__ b5,
    const float* __restrict__ Wf, const float* __restrict__ bf,
    float* __restrict__ out)
{
  extern __shared__ char smem_raw[];
  _Float16* xs = (_Float16*)smem_raw;          // [112][XP], adjacency f16 (resident all layers)
  _Float16* hT = xs + GNODE * XP;              // [256][HP], activations transposed
  _Float16* ag = hT + 256  * HP;               // [112][APCH], aggregation result
  float*   red = (float*)(ag + GNODE * APCH);  // [128] reduction scratch

  const int tid  = threadIdx.x;
  const int lane = tid & 31;
  const int wave = tid >> 5;
  const int b    = blockIdx.x;

  // Load x[b] once from HBM (vectorized float4) -> f16 LDS. 112 % 4 == 0 so each
  // float4 stays within one row; 4-half stores are 8B aligned.
  const float* xb = x + (size_t)b * (GNODE * GNODE);
  for (int i = tid; i < (GNODE * GNODE) / 4; i += 256) {
    v4f f = *(const v4f*)(xb + i * 4);
    int flat = i * 4;
    int n = flat / GNODE, m = flat - n * GNODE;
    v4h h = { (_Float16)f[0], (_Float16)f[1], (_Float16)f[2], (_Float16)f[3] };
    *(v4h*)(xs + n * XP + m) = h;
  }
  // zero K-pad cols 112..127 of xs (one 32B region per row) and of all 256 hT rows
  {
    v8h z = {};
    for (int n = tid; n < GNODE; n += 256) {
      *(v8h*)(xs + n * XP + GNODE)     = z;
      *(v8h*)(xs + n * XP + GNODE + 8) = z;
    }
    for (int d = tid; d < 256; d += 256) {
      *(v8h*)(hT + d * HP + GNODE)     = z;
      *(v8h*)(hT + d * HP + GNODE + 8) = z;
    }
  }
  __syncthreads();

  // Layer 1: h1 = relu(x @ W1^T + b1)   (first aggregation is identity)
  proj_stage(xs, XP, 128, W1p, b1, 256, hT, wave, lane);
  __syncthreads();
  agg_stage(xs, hT, ag, 256, wave, lane);                    // x @ h1
  __syncthreads();
  proj_stage(ag, APCH, 256, W2p, b2, 256, hT, wave, lane);   // layer 2
  __syncthreads();
  agg_stage(xs, hT, ag, 256, wave, lane);
  __syncthreads();
  proj_stage(ag, APCH, 256, W3p, b3, 128, hT, wave, lane);   // layer 3
  __syncthreads();
  agg_stage(xs, hT, ag, 128, wave, lane);
  __syncthreads();
  proj_stage(ag, APCH, 128, W4p, b4, 64, hT, wave, lane);    // layer 4
  __syncthreads();
  agg_stage(xs, hT, ag, 64, wave, lane);                     // x @ h4  -> [112][64]
  __syncthreads();

  // Layer 5 (D=1) + final head: out[b] = sum_n relu(ag[n]·W5 + b5) * Wf[n] + bf
  if (tid < GNODE) {
    float s = b5[0];
#pragma unroll
    for (int dv = 0; dv < 8; ++dv) {
      v8h av = *(const v8h*)(ag + tid * APCH + dv * 8);   // 16B aligned (APCH*2=528)
      v8h wv = *(const v8h*)(W5p + dv * 8);
#pragma unroll
      for (int j = 0; j < 8; ++j)
        s += (float)av[j] * (float)wv[j];
    }
    s = relu1(s);
    red[tid] = s * Wf[tid];
  } else if (tid < 128) {
    red[tid] = 0.0f;
  }
  __syncthreads();
  for (int st = 64; st > 0; st >>= 1) {
    if (tid < st) red[tid] += red[tid + st];
    __syncthreads();
  }
  if (tid == 0) out[b] = red[0] + bf[0];
}

extern "C" void kernel_launch(void* const* d_in, const int* in_sizes, int n_in,
                              void* d_out, int out_size, void* d_ws, size_t ws_size,
                              hipStream_t stream) {
  (void)in_sizes; (void)n_in; (void)out_size; (void)ws_size;
  const float* x  = (const float*)d_in[0];
  const float* W1 = (const float*)d_in[1];  const float* b1 = (const float*)d_in[2];
  const float* W2 = (const float*)d_in[3];  const float* b2 = (const float*)d_in[4];
  const float* W3 = (const float*)d_in[5];  const float* b3 = (const float*)d_in[6];
  const float* W4 = (const float*)d_in[7];  const float* b4 = (const float*)d_in[8];
  const float* W5 = (const float*)d_in[9];  const float* b5 = (const float*)d_in[10];
  const float* Wf = (const float*)d_in[11]; const float* bf = (const float*)d_in[12];
  float* out = (float*)d_out;

  // f16 padded weights in workspace
  _Float16* W1p = (_Float16*)d_ws;          // 256 x 128
  _Float16* W2p = W1p + 256 * 128;          // 256 x 256
  _Float16* W3p = W2p + 256 * 256;          // 128 x 256
  _Float16* W4p = W3p + 128 * 256;          //  64 x 128
  _Float16* W5p = W4p + 64 * 128;           //  64

  cvt_pad<<<(256 * 128 + 255) / 256, 256, 0, stream>>>(W1, W1p, 256, 112, 128);
  cvt_pad<<<(256 * 256 + 255) / 256, 256, 0, stream>>>(W2, W2p, 256, 256, 256);
  cvt_pad<<<(128 * 256 + 255) / 256, 256, 0, stream>>>(W3, W3p, 128, 256, 256);
  cvt_pad<<<( 64 * 128 + 255) / 256, 256, 0, stream>>>(W4, W4p,  64, 128, 128);
  cvt_pad<<<1, 256, 0, stream>>>(W5, W5p, 1, 64, 64);

  const int SMEM = (GNODE * XP + 256 * HP + GNODE * APCH) * 2 + 128 * (int)sizeof(float);
  hipFuncSetAttribute((const void*)gcn_main, hipFuncAttributeMaxDynamicSharedMemorySize, SMEM);
  gcn_main<<<4096, 256, SMEM, stream>>>(x, W1p, b1, W2p, b2, W3p, b3, W4p, b4,
                                        W5p, b5, Wf, bf, out);
}
